// RGCN_76493367542117
// MI455X (gfx1250) — compile-verified, matrix operand order checked by reading, not run on hardware
//
#include <hip/hip_runtime.h>
#include <hip/hip_bf16.h>

// ---------------- problem constants (match reference) ----------------
#define Nn   20000
#define Ee   320000
#define Rr   8
#define Dd   256
#define Ll   6
#define EEe  32
#define Bb   64
#define Kk   32
#define KTOT (Rr*Dd + Dd)   // 2304 = concat-K (8 relations * 256 + self 256)
#define KTILES (KTOT/32)    // 72 WMMA K-steps
#define NCH   (KTOT/128)    // 18 K-chunks of 128
#define MT    80            // GEMM M-tile rows (20000/80 = 250 blocks exactly)
#define MSUB  5             // 5 row-subtiles of 16 per wave

typedef __attribute__((ext_vector_type(16))) __bf16 v16bf;
typedef __attribute__((ext_vector_type(8)))  float  v8f;

union AF { v16bf v; uint4 q[2]; __bf16 e[16]; };

__device__ __forceinline__ float warp_sum(float v) {
#pragma unroll
  for (int o = 16; o; o >>= 1) v += __shfl_xor(v, o, 32);
  return v;
}

// ---------------- utility kernels ----------------
__global__ void k_fill4(float4* __restrict__ p, float v, size_t n4) {
  size_t i = (size_t)blockIdx.x * blockDim.x + threadIdx.x;
  if (i < n4) p[i] = make_float4(v, v, v, v);
}

__global__ void k_deg(const int* __restrict__ dst, float* __restrict__ deg) {
  int e = blockIdx.x * blockDim.x + threadIdx.x;
  if (e < Ee) unsafeAtomicAdd(&deg[dst[e]], 1.0f);
}

__global__ void k_deginv(const float* __restrict__ deg, float* __restrict__ dinv) {
  int n = blockIdx.x * blockDim.x + threadIdx.x;
  if (n < Nn) dinv[n] = 1.0f / fmaxf(deg[n], 1.0f);
}

// ---------------- per-layer weight conversion to WMMA-fragment order ----------------
// B fragment layout (32x16 bf16 per tile): lane = {half=lane>>4, n=lane&15},
// element e -> k = ktile*32 + ((e&8)?16:0) + half*8 + (e&7), col = ntile*16 + n.
// Linear order idx = ((kt*16 + nt)*32 + lane)*16 + e means one K-chunk of 4 tiles
// x 256 cols is a CONTIGUOUS 64KB block -> perfect for async copy to LDS.
__global__ void k_convW(const float* __restrict__ Wl, const float* __restrict__ Wsl,
                        __bf16* __restrict__ Wcat) {
  int idx = blockIdx.x * blockDim.x + threadIdx.x;
  if (idx >= KTILES * 16 * 32 * 16) return;
  int e    = idx & 15;
  int lane = (idx >> 4) & 31;
  int nt   = (idx >> 9) & 15;
  int kt   = idx >> 13;
  int half = lane >> 4, n = lane & 15;
  int kk   = kt * 32 + ((e & 8) ? 16 : 0) + half * 8 + (e & 7);
  int col  = nt * 16 + n;
  float v = (kk < Rr * Dd) ? Wl[((kk >> 8) * Dd + (kk & 255)) * Dd + col]
                           : Wsl[(kk & 255) * Dd + col];
  Wcat[idx] = (__bf16)v;
}

__global__ void k_convWe(const float* __restrict__ Wel, __bf16* __restrict__ Wesw) {
  int idx = blockIdx.x * blockDim.x + threadIdx.x;
  if (idx >= 16 * 32 * 16) return;
  int e    = idx & 15;
  int lane = (idx >> 4) & 31;
  int nt   = (idx >> 9) & 15;
  int half = lane >> 4, n = lane & 15;
  int kk   = ((e & 8) ? 16 : 0) + half * 8 + (e & 7);   // K = 0..31 (single tile)
  Wesw[idx] = (__bf16)Wel[kk * Dd + nt * 16 + n];
}

// ---------------- scatter: msg = x[src] + edge_embed @ We ; agg[rel*N+dst] += msg ----
// All agg offsets fit in 32 bits (max 40.96M elements) -> saddr+voffset atomics.
__global__ void __launch_bounds__(512)
k_scatter(const float* __restrict__ x, const float* __restrict__ ee,
          const int* __restrict__ ei, const int* __restrict__ etype,
          const __bf16* __restrict__ Wesw, float* __restrict__ agg) {
  const int tid  = threadIdx.x;
  const int w    = tid >> 5;       // wave 0..15  -> N-tile
  const int lane = tid & 31;
  const int half = lane >> 4, n = lane & 15;
  const int base = blockIdx.x * 16;       // E = 20000 tiles of 16 edges exactly

  // A fragment: 16 edges x 32 EE dims (bf16)
  AF a;
  {
    const int em = base + n;
    const float* p0 = &ee[em * EEe + half * 8];
#pragma unroll
    for (int i = 0; i < 8; ++i) {
      a.e[i]     = (__bf16)p0[i];
      a.e[8 + i] = (__bf16)p0[16 + i];
    }
  }
  AF b;
  {
    const uint4* pb = (const uint4*)&Wesw[(w * 32 + lane) * 16];
    b.q[0] = pb[0]; b.q[1] = pb[1];
  }
  v8f c;
#pragma unroll
  for (int i = 0; i < 8; ++i) c[i] = 0.0f;
  c = __builtin_amdgcn_wmma_f32_16x16x32_bf16(false, a.v, false, b.v,
                                              (short)0, c, false, false);
#pragma unroll
  for (int v = 0; v < 8; ++v) {
    const int m = v + half * 8;
    const int e = base + m;
    const int s = ei[e];          // src
    const int d = ei[Ee + e];     // dst
    const int r = etype[e];
    const int col = w * 16 + n;
    const float val = c[v] + x[(unsigned)(s * Dd + col)];
    const unsigned aidx = (unsigned)(r * Nn + d) * Dd + (unsigned)col;  // < 2^31
    unsafeAtomicAdd(agg + aidx, val);
  }
}

// ---------------- fused concat-K GEMM + LayerNorm + ReLU + residual --------------
// Block: 512 threads (16 waves). Tile: 80 node rows x 256 cols, K = 2304.
// - A (agg*1/deg | x) staged in LDS as bf16, 128-K chunks, fp32->bf16 on the fly.
// - B (Wcat, fragment-ordered bf16) double-buffered in LDS via
//   global_load_async_to_lds_b128 (ASYNCcnt), issued one chunk ahead.
// - Each wave: 1 B fragment feeds 5 row-subtile WMMAs (5 accumulators).
// - Epilogue: per-row LayerNorm + ReLU + residual, in place into x.
__global__ void __launch_bounds__(512)
k_gemm_ln(const float* __restrict__ agg, float* __restrict__ x,
          const float* __restrict__ dinv, const __bf16* __restrict__ Wcat,
          const float* __restrict__ g, const float* __restrict__ bta) {
  // LDS: As 20KB | Bs 2x64KB (Hs 80KB aliases Bs after the K-loop)
  __shared__ __align__(16) char smem[MT * 128 * 2 + 2 * 65536];
  __bf16* As = (__bf16*)smem;
  char*   BsBase = smem + MT * 128 * 2;
  float*  Hs = (float*)BsBase;

  const int tid  = threadIdx.x;
  const int w    = tid >> 5;
  const int lane = tid & 31;
  const int half = lane >> 4, n = lane & 15;
  const int rowbase = blockIdx.x * MT;

  v8f c[MSUB];
#pragma unroll
  for (int m0 = 0; m0 < MSUB; ++m0)
#pragma unroll
    for (int i = 0; i < 8; ++i) c[m0][i] = 0.0f;

  const unsigned lds0 = (unsigned)(uintptr_t)BsBase;  // LDS addr = low 32 bits of flat

  // prefetch B chunk 0 into buffer 0 (8 x b128 per thread-slice = 64KB/block)
  {
    const char* gp = (const char*)Wcat + (size_t)tid * 16;
    unsigned lds = lds0 + tid * 16u;
#pragma unroll
    for (int i = 0; i < 8; ++i) {
      asm volatile("global_load_async_to_lds_b128 %0, %1, off"
                   :: "v"(lds + i * 8192u), "v"(gp + i * 8192) : "memory");
    }
  }

  for (int ch = 0; ch < NCH; ++ch) {
    // cooperative A-chunk load: 80 rows x 128 K, fp32 -> bf16 (+1/deg scaling)
#pragma unroll
    for (int t = 0; t < MT * 128 / 512; ++t) {
      const int j    = tid + t * 512;
      const int r    = j >> 7;              // row 0..79
      const int k    = j & 127;
      const int kk   = ch * 128 + k;
      const int node = rowbase + r;
      float v;
      if (kk < Rr * Dd)   // 32-bit exact offsets (< 2^31)
        v = agg[(unsigned)((kk >> 8) * Nn + node) * Dd + (unsigned)(kk & 255)]
            * dinv[node];
      else
        v = x[(unsigned)(node * Dd + (kk & 255))];
      As[r * 128 + k] = (__bf16)v;
    }

    // issue async copy of NEXT B chunk into the other buffer
    if (ch + 1 < NCH) {
      const char* gp = (const char*)Wcat + (size_t)(ch + 1) * 65536 + (size_t)tid * 16;
      unsigned lds = lds0 + (unsigned)(((ch + 1) & 1) * 65536) + tid * 16u;
#pragma unroll
      for (int i = 0; i < 8; ++i) {
        asm volatile("global_load_async_to_lds_b128 %0, %1, off"
                     :: "v"(lds + i * 8192u), "v"(gp + i * 8192) : "memory");
      }
      asm volatile("s_wait_asynccnt 0x8" ::: "memory");  // current chunk's 8 done
    } else {
      asm volatile("s_wait_asynccnt 0x0" ::: "memory");
    }
    __syncthreads();   // As ready + every wave's async portion of this chunk landed

    const __bf16* Bs = (const __bf16*)(BsBase + (ch & 1) * 65536);
#pragma unroll
    for (int s = 0; s < 4; ++s) {
      AF b;   // one B fragment per wave, reused by 5 row-subtiles
      const uint4* pb = (const uint4*)&Bs[((s * 16 + w) * 32 + lane) * 16];
      b.q[0] = pb[0]; b.q[1] = pb[1];
#pragma unroll
      for (int m0 = 0; m0 < MSUB; ++m0) {
        AF a;
        const uint4* pa = (const uint4*)&As[(m0 * 16 + n) * 128 + s * 32 + half * 8];
        a.q[0] = pa[0];      // K = kbase + half*8 .. +7
        a.q[1] = pa[2];      // K = kbase + 16 + half*8 .. +7 (skip 16B)
        c[m0] = __builtin_amdgcn_wmma_f32_16x16x32_bf16(false, a.v, false, b.v,
                                                        (short)0, c[m0], false, false);
      }
    }
    __syncthreads();   // done reading As and Bs[cur] before they are overwritten
  }

  // write C tiles to Hs (aliases Bs; safe after trailing barrier above)
#pragma unroll
  for (int m0 = 0; m0 < MSUB; ++m0)
#pragma unroll
    for (int v = 0; v < 8; ++v)
      Hs[(m0 * 16 + v + half * 8) * Dd + w * 16 + n] = c[m0][v];
  __syncthreads();

  // LayerNorm + ReLU + residual: wave w handles rows {w, 16+w, 32+w, 48+w, 64+w}
#pragma unroll
  for (int rr = 0; rr < MSUB; ++rr) {
    const int row  = rr * 16 + w;
    const int node = rowbase + row;
    float vals[8], s1 = 0.f, s2 = 0.f;
#pragma unroll
    for (int i = 0; i < 8; ++i) {
      const float hv = Hs[row * Dd + lane + i * 32];
      vals[i] = hv; s1 += hv; s2 += hv * hv;
    }
    s1 = warp_sum(s1); s2 = warp_sum(s2);
    const float mu  = s1 * (1.0f / Dd);
    const float var = s2 * (1.0f / Dd) - mu * mu;
    const float rs  = rsqrtf(var + 1e-5f);
#pragma unroll
    for (int i = 0; i < 8; ++i) {
      const int col = lane + i * 32;
      float y = (vals[i] - mu) * rs * g[col] + bta[col];
      y = fmaxf(y, 0.0f);
      x[node * Dd + col] = y + x[node * Dd + col];   // residual, in place
    }
  }
}

// ---------------- DistMult scoring: out[b,k] = sum_d x[s]*rel[r]*x[t] -------------
__global__ void k_score(const float* __restrict__ x, const float* __restrict__ rel,
                        const int* __restrict__ batch, float* __restrict__ out) {
  const int wid  = (blockIdx.x * blockDim.x + threadIdx.x) >> 5;
  const int lane = threadIdx.x & 31;
  if (wid >= Bb * Kk) return;
  const int s = batch[wid * 3 + 0];
  const int t = batch[wid * 3 + 1];
  const int r = batch[wid * 3 + 2];
  float acc = 0.f;
#pragma unroll
  for (int i = 0; i < Dd / 32; ++i) {
    const int d = lane + i * 32;
    acc += x[s * Dd + d] * rel[r * Dd + d] * x[t * Dd + d];
  }
  acc = warp_sum(acc);
  if (lane == 0) out[wid] = acc;
}

// ---------------- host-side launcher ----------------
extern "C" void kernel_launch(void* const* d_in, const int* in_sizes, int n_in,
                              void* d_out, int out_size, void* d_ws, size_t ws_size,
                              hipStream_t stream) {
  (void)in_sizes; (void)n_in; (void)out_size; (void)ws_size;
  const int*   edge_index = (const int*)  d_in[0];   // (2,E)
  const int*   edge_type  = (const int*)  d_in[1];   // (E,)
  const float* edge_embed = (const float*)d_in[2];   // (E,EE)
  const int*   batch      = (const int*)  d_in[3];   // (B,K,3)
  const float* W          = (const float*)d_in[4];   // (L,R,D,D)
  const float* Wself      = (const float*)d_in[5];   // (L,D,D)
  const float* We         = (const float*)d_in[6];   // (L,EE,D)
  const float* gamma      = (const float*)d_in[7];   // (L,D)
  const float* beta       = (const float*)d_in[8];   // (L,D)
  const float* rel_emb    = (const float*)d_in[9];   // (R,D)
  float* out = (float*)d_out;

  // workspace layout
  char* p = (char*)d_ws;
  float* x    = (float*)p;  p += (size_t)Nn * Dd * 4;          // 20.5 MB
  float* agg  = (float*)p;  p += (size_t)Rr * Nn * Dd * 4;     // 163.8 MB (L2-resident)
  float* deg  = (float*)p;  p += (size_t)Nn * 4;
  float* dinv = (float*)p;  p += (size_t)Nn * 4;
  __bf16* Wcat = (__bf16*)p; p += (size_t)KTOT * Dd * 2;       // 1.2 MB
  __bf16* Wesw = (__bf16*)p;                                   // 16 KB

  const size_t nx   = (size_t)Nn * Dd;
  const size_t nagg = (size_t)Rr * Nn * Dd;

  k_fill4<<<(unsigned)((nx / 4 + 255) / 256), 256, 0, stream>>>((float4*)x, 1.0f, nx / 4);
  k_fill4<<<(Nn / 4 + 255) / 256, 256, 0, stream>>>((float4*)deg, 0.0f, Nn / 4);
  k_deg<<<(Ee + 255) / 256, 256, 0, stream>>>(edge_index + Ee, deg);
  k_deginv<<<(Nn + 255) / 256, 256, 0, stream>>>(deg, dinv);

  for (int l = 0; l < Ll; ++l) {
    k_fill4<<<(unsigned)((nagg / 4 + 255) / 256), 256, 0, stream>>>((float4*)agg, 0.0f,
                                                                    nagg / 4);
    k_convW<<<(KTILES * 16 * 32 * 16 + 255) / 256, 256, 0, stream>>>(
        W + (size_t)l * Rr * Dd * Dd, Wself + (size_t)l * Dd * Dd, Wcat);
    k_convWe<<<(16 * 32 * 16 + 255) / 256, 256, 0, stream>>>(
        We + (size_t)l * EEe * Dd, Wesw);
    k_scatter<<<Ee / 16, 512, 0, stream>>>(x, edge_embed, edge_index, edge_type,
                                           Wesw, agg);
    k_gemm_ln<<<Nn / MT, 512, 0, stream>>>(agg, x, dinv, Wcat,
                                           gamma + (size_t)l * Dd,
                                           beta + (size_t)l * Dd);
  }
  k_score<<<(Bb * Kk * 32 + 255) / 256, 256, 0, stream>>>(x, rel_emb, batch, out);
}
